// MoE_85315230368399
// MI455X (gfx1250) — matrix operand order repaired and checked by reading, not compiled, not run on hardware
//
#include <hip/hip_runtime.h>

typedef __bf16 bf16;
typedef bf16  v16bf __attribute__((ext_vector_type(16)));
typedef bf16  v8bf  __attribute__((ext_vector_type(8)));
typedef bf16  v2bf  __attribute__((ext_vector_type(2)));
typedef float v8f   __attribute__((ext_vector_type(8)));

constexpr int kT  = 8192;   // B*S tokens
constexpr int kH  = 1024;   // hidden
constexpr int kE  = 16;     // experts
constexpr int kI  = 4096;   // gate intermediate
constexpr int kIH = 2048;   // I/2
constexpr int TM  = 64;     // token tile
constexpr int TN  = 64;     // column tile
constexpr int KC  = 32;     // K step (matches 16x16x32 bf16 WMMA)

#if __has_builtin(__builtin_amdgcn_global_load_async_to_lds_b128)
#define USE_ASYNC_LDS 1
#else
#define USE_ASYNC_LDS 0
#endif

#if USE_ASYNC_LDS
typedef int v4i_vs __attribute__((vector_size(4 * sizeof(int))));
typedef __attribute__((address_space(1))) v4i_vs* gptr_v4i;
typedef __attribute__((address_space(3))) v4i_vs* lptr_v4i;
__device__ __forceinline__ void async_copy_b128(const bf16* g, bf16* l) {
  __builtin_amdgcn_global_load_async_to_lds_b128((gptr_v4i)g, (lptr_v4i)l, 0, 0);
}
__device__ __forceinline__ void wait_async0() {
#if __has_builtin(__builtin_amdgcn_s_wait_asynccnt)
  __builtin_amdgcn_s_wait_asynccnt(0);
#else
  asm volatile("s_wait_asynccnt 0" ::: "memory");
#endif
}
#endif

__device__ __forceinline__ v8f wmma_bf16(v16bf a, v16bf b, v8f c) {
  return __builtin_amdgcn_wmma_f32_16x16x32_bf16(false, a, false, b, (short)0, c,
                                                 false, false);
}

// A fragment: 16x32 bf16, lA row-major [rows][KC].
// lanes 0-15: M=lane, K = 0..7 & 16..23 ; lanes 16-31: M=lane-16, K = 8..15 & 24..31
__device__ __forceinline__ v16bf load_a_frag(const bf16* lA, int mBase, int lane) {
  const bf16* p = lA + (mBase + (lane & 15)) * KC;
  const int k0 = (lane >> 4) * 8;
  v16bf a;
#pragma unroll
  for (int i = 0; i < 8; ++i) { a[i] = p[k0 + i]; a[8 + i] = p[16 + k0 + i]; }
  return a;
}

// B fragment: 32x16 bf16, lB stored TRANSPOSED [N][KC].
// lanes 0-15: N=lane, K = 0..7 & 16..23 ; lanes 16-31: N=lane-16, K = 8..15 & 24..31
__device__ __forceinline__ v16bf load_b_frag(const bf16* lB, int nBase, int lane) {
  const bf16* p = lB + (nBase + (lane & 15)) * KC;
  const int k0 = (lane >> 4) * 8;
  v16bf b;
#pragma unroll
  for (int i = 0; i < 8; ++i) { b[i] = p[k0 + i]; b[8 + i] = p[16 + k0 + i]; }
  return b;
}

__device__ __forceinline__ v2bf pk2(float lo, float hi) {
  v2bf r; r[0] = (bf16)lo; r[1] = (bf16)hi; return r;
}

__device__ __forceinline__ v8bf pk8(float4 a, float4 b) {
  v8bf r;
  r[0] = (bf16)a.x; r[1] = (bf16)a.y; r[2] = (bf16)a.z; r[3] = (bf16)a.w;
  r[4] = (bf16)b.x; r[5] = (bf16)b.y; r[6] = (bf16)b.z; r[7] = (bf16)b.w;
  return r;
}

// ---------------------------------------------------------------- zero init
__global__ void k_zero(float* __restrict__ out, int n, int* __restrict__ counts) {
  int i = blockIdx.x * blockDim.x + threadIdx.x;
  if (i < n) out[i] = 0.0f;
  if (blockIdx.x == 0 && threadIdx.x < kE) counts[threadIdx.x] = 0;
}

// ---------------------------------------------------------------- router
__global__ void k_router(const float* __restrict__ x, const float* __restrict__ rw,
                         int* __restrict__ topk_i, float* __restrict__ topk_w,
                         int* __restrict__ counts) {
  const int wave = (blockIdx.x * blockDim.x + threadIdx.x) >> 5;
  const int lane = threadIdx.x & 31;
  if (wave >= kT) return;
  const float* xr = x + (size_t)wave * kH;
  float acc[kE];
#pragma unroll
  for (int e = 0; e < kE; ++e) acc[e] = 0.0f;
  for (int h = lane; h < kH; h += 32) {
    const float xv = xr[h];
    const float* w = rw + h * kE;
#pragma unroll
    for (int e = 0; e < kE; ++e) acc[e] += xv * w[e];
  }
#pragma unroll
  for (int e = 0; e < kE; ++e) {
#pragma unroll
    for (int off = 16; off >= 1; off >>= 1) acc[e] += __shfl_xor(acc[e], off, 32);
  }
  if (lane == 0) {
    int i1 = 0, i2 = -1;
    float l1 = -3.4e38f, l2 = -3.4e38f;
#pragma unroll
    for (int e = 0; e < kE; ++e) {
      const float v = acc[e];
      if (v > l1) { l2 = l1; i2 = i1; l1 = v; i1 = e; }
      else if (v > l2) { l2 = v; i2 = e; }
    }
    const float e2 = __expf(l2 - l1);
    const float inv = __builtin_amdgcn_rcpf(1.0f + e2);
    topk_i[2 * wave + 0] = i1;
    topk_i[2 * wave + 1] = i2;
    topk_w[2 * wave + 0] = inv;
    topk_w[2 * wave + 1] = e2 * inv;
    atomicAdd(&counts[i1], 1);
    atomicAdd(&counts[i2], 1);
  }
}

// ---------------------------------------------------------------- offsets
__global__ void k_offsets(const int* __restrict__ counts, int* __restrict__ offsets) {
  if (threadIdx.x == 0) {
    int s = 0;
    for (int e = 0; e < kE; ++e) { offsets[e] = s; s += counts[e]; }
    offsets[kE] = s;
  }
}

// ---------------------------------------------------------------- scatter
// deterministic stream compaction: block e gathers tokens routed to expert e
__global__ void k_scatter(const int* __restrict__ tki, const float* __restrict__ tkw,
                          const int* __restrict__ offsets, int* __restrict__ tok_list,
                          float* __restrict__ tok_w) {
  const int e = blockIdx.x;
  __shared__ int wsum[8];
  __shared__ int chunkTot;
  const int lane = threadIdx.x & 31, wv = threadIdx.x >> 5;
  int running = offsets[e];
  for (int t0 = 0; t0 < kT; t0 += 256) {
    const int t = t0 + threadIdx.x;
    bool take = false; float w = 0.0f;
    const int a = tki[2 * t], b = tki[2 * t + 1];
    if (a == e)      { take = true; w = tkw[2 * t]; }
    else if (b == e) { take = true; w = tkw[2 * t + 1]; }
    const unsigned long long m = __ballot(take);
    const int pre = __popcll(m & ((1ull << lane) - 1ull));
    if (lane == 0) wsum[wv] = __popcll(m);
    __syncthreads();
    if (threadIdx.x == 0) {
      int s = 0;
      for (int i = 0; i < 8; ++i) { const int c = wsum[i]; wsum[i] = s; s += c; }
      chunkTot = s;
    }
    __syncthreads();
    if (take) {
      const int pos = running + wsum[wv] + pre;
      tok_list[pos] = t;
      tok_w[pos] = w;
    }
    running += chunkTot;
    __syncthreads();
  }
}

// ---------------------------------------------------------------- gate/up GEMM + SiLU
// h[slot, n] = silu(x@gate[:, n+IH]) * (x@gate[:, n]) * (x@up[:, n])
__global__ void __launch_bounds__(256)
k_expert_up(const float* __restrict__ x, const float* __restrict__ gate_w,
            const float* __restrict__ up_w, const int* __restrict__ counts,
            const int* __restrict__ offsets, const int* __restrict__ tok_list,
            bf16* __restrict__ hbuf) {
  const int e = blockIdx.z;
  const int cnt = counts[e];
  const int rowBase = blockIdx.y * TM;
  if (rowBase >= cnt) return;
  const int off = offsets[e];
  const int n0 = blockIdx.x * TN;
  const float* gw = gate_w + (size_t)e * kH * kI;
  const float* uw = up_w   + (size_t)e * kH * kIH;

  __shared__ bf16 lA [TM][KC];
  __shared__ bf16 lB1[TN][KC];   // gate cols n0..        [N][K] transposed
  __shared__ bf16 lB2[TN][KC];   // gate cols n0+IH..     [N][K]
  __shared__ bf16 lBu[TN][KC];   // up   cols n0..        [N][K]
  __shared__ int  lTok[TM];

  const int tid = threadIdx.x;
  if (tid < TM) {
    int r = rowBase + tid;
    if (r > cnt - 1) r = cnt - 1;
    lTok[tid] = tok_list[off + r];
  }
  __syncthreads();

  const int lane = tid & 31, wv = tid >> 5;
  const int mb  = (wv >> 1) * 16;   // wave's row block
  const int nb0 = (wv & 1) * 32;    // wave's first col block (owns nb0, nb0+16)

  // loader geometry
  const int ar = tid >> 2, ac = (tid & 3) * 8;          // A: row, 8 K-cols
  const int br = (tid >> 4) * 2;                        // B: even K-row of pair
  const int bc = (tid & 15) * 4;                        // B: 4 N-cols

  const float* aBase  = x + (size_t)lTok[ar] * kH + ac;
  float4 ra0, ra1, rg1a, rg1b, rg2a, rg2b, rua, rub;

  auto load_stage = [&](int kk) {
    const float* pa = aBase + kk;
    ra0 = *(const float4*)(pa);
    ra1 = *(const float4*)(pa + 4);
    const float* p1 = gw + (size_t)(kk + br) * kI + n0 + bc;
    rg1a = *(const float4*)(p1);
    rg1b = *(const float4*)(p1 + kI);
    const float* p2 = p1 + kIH;
    rg2a = *(const float4*)(p2);
    rg2b = *(const float4*)(p2 + kI);
    const float* pu = uw + (size_t)(kk + br) * kIH + n0 + bc;
    rua = *(const float4*)(pu);
    rub = *(const float4*)(pu + kIH);
  };

  const v8f vzero = {0.f, 0.f, 0.f, 0.f, 0.f, 0.f, 0.f, 0.f};
  v8f ag1[2] = {vzero, vzero};
  v8f ag2[2] = {vzero, vzero};
  v8f au [2] = {vzero, vzero};

  load_stage(0);
  for (int kk = 0; kk < kH; kk += KC) {
    // commit staged registers to LDS (A packed b128; B packed K-pairs -> b32)
    *(v8bf*)&lA[ar][ac] = pk8(ra0, ra1);
    const float g1lo[4] = {rg1a.x, rg1a.y, rg1a.z, rg1a.w};
    const float g1hi[4] = {rg1b.x, rg1b.y, rg1b.z, rg1b.w};
    const float g2lo[4] = {rg2a.x, rg2a.y, rg2a.z, rg2a.w};
    const float g2hi[4] = {rg2b.x, rg2b.y, rg2b.z, rg2b.w};
    const float uplo[4] = {rua.x, rua.y, rua.z, rua.w};
    const float uphi[4] = {rub.x, rub.y, rub.z, rub.w};
#pragma unroll
    for (int j = 0; j < 4; ++j) {
      *(v2bf*)&lB1[bc + j][br] = pk2(g1lo[j], g1hi[j]);
      *(v2bf*)&lB2[bc + j][br] = pk2(g2lo[j], g2hi[j]);
      *(v2bf*)&lBu[bc + j][br] = pk2(uplo[j], uphi[j]);
    }
    __syncthreads();

    // prefetch next K-step while WMMAs run
    if (kk + KC < kH) load_stage(kk + KC);

    const v16bf a   = load_a_frag(&lA [0][0], mb, lane);
    const v16bf b10 = load_b_frag(&lB1[0][0], nb0,      lane);
    const v16bf b11 = load_b_frag(&lB1[0][0], nb0 + 16, lane);
    const v16bf b20 = load_b_frag(&lB2[0][0], nb0,      lane);
    const v16bf b21 = load_b_frag(&lB2[0][0], nb0 + 16, lane);
    const v16bf bu0 = load_b_frag(&lBu[0][0], nb0,      lane);
    const v16bf bu1 = load_b_frag(&lBu[0][0], nb0 + 16, lane);
    ag1[0] = wmma_bf16(a, b10, ag1[0]);
    ag1[1] = wmma_bf16(a, b11, ag1[1]);
    ag2[0] = wmma_bf16(a, b20, ag2[0]);
    ag2[1] = wmma_bf16(a, b21, ag2[1]);
    au [0] = wmma_bf16(a, bu0, au [0]);
    au [1] = wmma_bf16(a, bu1, au [1]);
    __syncthreads();
  }

  // epilogue: h = silu(g2) * g1 * up, store bf16 (fast rcp instead of IEEE div)
  const int nloc = lane & 15;
  const int mhi  = (lane >> 4) * 8;
#pragma unroll
  for (int j = 0; j < 2; ++j) {
    const int col = n0 + nb0 + j * 16 + nloc;
#pragma unroll
    for (int v = 0; v < 8; ++v) {
      const int r = rowBase + mb + mhi + v;
      if (r < cnt) {
        const float g1 = ag1[j][v], g2 = ag2[j][v], up = au[j][v];
        const float sig = __builtin_amdgcn_rcpf(1.0f + __expf(-g2));
        const float h = g2 * sig * g1 * up;
        hbuf[(size_t)(off + r) * kIH + col] = (bf16)h;
      }
    }
  }
}

// ---------------------------------------------------------------- down GEMM + scatter
// A tile (bf16 h rows) uses async global->LDS DMA with double buffering;
// B tile needs fp32->bf16 conversion so it stays on the register-staged path.
__global__ void __launch_bounds__(256)
k_expert_down(const bf16* __restrict__ hbuf, const float* __restrict__ down_w,
              const int* __restrict__ counts, const int* __restrict__ offsets,
              const int* __restrict__ tok_list, const float* __restrict__ tok_w,
              float* __restrict__ out) {
  const int e = blockIdx.z;
  const int cnt = counts[e];
  const int rowBase = blockIdx.y * TM;
  if (rowBase >= cnt) return;
  const int off = offsets[e];
  const int n0 = blockIdx.x * TN;
  const float* dw = down_w + (size_t)e * kIH * kH;

  __shared__ bf16  lA[2][TM][KC];   // double-buffered async dest
  __shared__ bf16  lB[TN][KC];      // down cols, [N][K] transposed
  __shared__ int   lSlot[TM];
  __shared__ int   lTok [TM];
  __shared__ float lW   [TM];

  const int tid = threadIdx.x;
  if (tid < TM) {
    int r = rowBase + tid;
    const bool valid = r < cnt;
    if (r > cnt - 1) r = cnt - 1;
    lSlot[tid] = off + r;
    lTok [tid] = tok_list[off + r];
    lW   [tid] = valid ? tok_w[off + r] : 0.0f;
  }
  __syncthreads();

  const int lane = tid & 31, wv = tid >> 5;
  const int mb  = (wv >> 1) * 16;
  const int nb0 = (wv & 1) * 32;

  const int ar = tid >> 2, ac = (tid & 3) * 8;
  const int br = (tid >> 4) * 2;
  const int bc = (tid & 15) * 4;

  const bf16* aBase = hbuf + (size_t)lSlot[ar] * kIH + ac;
  float4 rb0, rb1;
#if !USE_ASYNC_LDS
  uint4 rAv;
#endif

  auto load_b_stage = [&](int kk) {
    const float* p = dw + (size_t)(kk + br) * kH + n0 + bc;
    rb0 = *(const float4*)(p);
    rb1 = *(const float4*)(p + kH);
  };

  const v8f vzero = {0.f, 0.f, 0.f, 0.f, 0.f, 0.f, 0.f, 0.f};
  v8f acc[2] = {vzero, vzero};

  int cur = 0;
#if USE_ASYNC_LDS
  async_copy_b128(aBase, &lA[0][ar][ac]);   // A(kk=0) in flight
#else
  rAv = *(const uint4*)(aBase);
#endif
  load_b_stage(0);

  for (int kk = 0; kk < kIH; kk += KC) {
    // commit B tile; A tile arrives via async DMA (or staged reg on fallback)
#if !USE_ASYNC_LDS
    *(uint4*)&lA[cur][ar][ac] = rAv;
#endif
    const float blo[4] = {rb0.x, rb0.y, rb0.z, rb0.w};
    const float bhi[4] = {rb1.x, rb1.y, rb1.z, rb1.w};
#pragma unroll
    for (int j = 0; j < 4; ++j) *(v2bf*)&lB[bc + j][br] = pk2(blo[j], bhi[j]);
#if USE_ASYNC_LDS
    wait_async0();                          // this wave's A copy landed
#endif
    __syncthreads();

    if (kk + KC < kIH) {
#if USE_ASYNC_LDS
      async_copy_b128(aBase + kk + KC, &lA[cur ^ 1][ar][ac]);  // overlaps WMMAs
#else
      rAv = *(const uint4*)(aBase + kk + KC);
#endif
      load_b_stage(kk + KC);
    }

    const v16bf a  = load_a_frag(&lA[cur][0][0], mb, lane);
    const v16bf b0 = load_b_frag(&lB[0][0], nb0,      lane);
    const v16bf b1 = load_b_frag(&lB[0][0], nb0 + 16, lane);
    acc[0] = wmma_bf16(a, b0, acc[0]);
    acc[1] = wmma_bf16(a, b1, acc[1]);
    __syncthreads();
    cur ^= 1;
  }

  const int nloc = lane & 15;
  const int mhi  = (lane >> 4) * 8;
#pragma unroll
  for (int j = 0; j < 2; ++j) {
    const int col = n0 + nb0 + j * 16 + nloc;
#pragma unroll
    for (int v = 0; v < 8; ++v) {
      const int mrow = mb + mhi + v;
      if (rowBase + mrow < cnt) {
        const float val = acc[j][v] * lW[mrow];
        unsafeAtomicAdd(&out[(size_t)lTok[mrow] * kH + col], val);
      }
    }
  }
}

// ---------------------------------------------------------------- launch
extern "C" void kernel_launch(void* const* d_in, const int* in_sizes, int n_in,
                              void* d_out, int out_size, void* d_ws, size_t ws_size,
                              hipStream_t stream) {
  const float* x  = (const float*)d_in[0];   // [T, H]
  const float* rw = (const float*)d_in[1];   // [H, E]
  const float* gw = (const float*)d_in[2];   // [E, H, I]
  const float* uw = (const float*)d_in[3];   // [E, H, I/2]
  const float* dw = (const float*)d_in[4];   // [E, I/2, H]
  float* out = (float*)d_out;                // [T, H]

  char*  ws = (char*)d_ws;
  size_t p  = 0;
  auto alloc = [&](size_t bytes) -> void* {
    void* r = ws + p;
    p = (p + bytes + 255) & ~(size_t)255;
    return r;
  };
  int*   topk_i   = (int*)  alloc((size_t)kT * 2 * sizeof(int));
  float* topk_w   = (float*)alloc((size_t)kT * 2 * sizeof(float));
  int*   counts   = (int*)  alloc(kE * sizeof(int));
  int*   offsets  = (int*)  alloc((kE + 1) * sizeof(int));
  int*   tok_list = (int*)  alloc((size_t)kT * 2 * sizeof(int));
  float* tok_wv   = (float*)alloc((size_t)kT * 2 * sizeof(float));
  bf16*  hbuf     = (bf16*) alloc((size_t)kT * 2 * kIH * sizeof(bf16));  // 64 MB

  k_zero<<<(kT * kH + 255) / 256, 256, 0, stream>>>(out, kT * kH, counts);
  k_router<<<kT / 8, 256, 0, stream>>>(x, rw, topk_i, topk_w, counts);
  k_offsets<<<1, 32, 0, stream>>>(counts, offsets);
  k_scatter<<<kE, 256, 0, stream>>>(topk_i, topk_w, offsets, tok_list, tok_wv);
  k_expert_up<<<dim3(kIH / TN, kT / TM, kE), 256, 0, stream>>>(
      x, gw, uw, counts, offsets, tok_list, hbuf);
  k_expert_down<<<dim3(kH / TN, kT / TM, kE), 256, 0, stream>>>(
      hbuf, dw, counts, offsets, tok_list, tok_wv, out);
}
